// SimpleRGCN_11519102288003
// MI455X (gfx1250) — matrix operand order repaired and checked by reading, not compiled, not run on hardware
//
#include <hip/hip_runtime.h>
#include <hip/hip_bf16.h>

// ---------------------------------------------------------------------------
// RGCN on gfx1250 (MI455X): sort edges by relation so every 16-edge tile is a
// dense 16xK @ KxN GEMM on v_wmma_f32_16x16x32_f16 (f16 in, f32 accumulate),
// scatter results with global_atomic_add_f32. Problem is memory/atomic bound
// (~2 GB of traffic @ 23.3 TB/s); WMMA keeps the 84 GFLOP of matmul far off
// the critical path (VALU fp32 would dominate at ~0.5 ms).
// ---------------------------------------------------------------------------

typedef _Float16 v16h __attribute__((ext_vector_type(16)));
typedef _Float16 v8h  __attribute__((ext_vector_type(8)));
typedef float    v8f  __attribute__((ext_vector_type(8)));

#define EMB_DIM 128
#define HID_DIM 256
#define NREL    16

// ---------------- utility fills ----------------
__global__ void fill_u32_kernel(unsigned* __restrict__ p, unsigned v, size_t n) {
    size_t i = (size_t)blockIdx.x * blockDim.x + threadIdx.x;
    size_t stride = (size_t)gridDim.x * blockDim.x;
    for (; i < n; i += stride) p[i] = v;
}

// ---------------- x -> f16 (with node_ids gather) ----------------
__global__ void cvt_x_kernel(const float* __restrict__ emb,
                             const int* __restrict__ node_ids,
                             _Float16* __restrict__ Xh, int total) {
    int i = blockIdx.x * blockDim.x + threadIdx.x;
    if (i >= total) return;
    int row = i >> 7, c = i & 127;
    Xh[i] = (_Float16)emb[(size_t)node_ids[row] * EMB_DIM + c];
}

// ---------------- pack weights into B-fragment layout ----------------
// Fragment frag = (r*KT + kt)*NT + nt holds 512 f16: [lane 0..31][j 0..15]
// with k = kt*32 + (lane>>4)*16 + j, n = nt*16 + (lane&15).
// Each lane then loads its 32 bytes fully coalesced in the GEMM.
template <int KDIM, int NOUT>
__global__ void pack_w_kernel(const float* __restrict__ W, _Float16* __restrict__ Wp) {
    constexpr int KT = KDIM / 32, NT = NOUT / 16;
    constexpr int TOTAL = NREL * KT * NT * 512;
    int i = blockIdx.x * blockDim.x + threadIdx.x;
    if (i >= TOTAL) return;
    int j    = i & 15;
    int lane = (i >> 4) & 31;
    int frag = i >> 9;
    int nt   = frag % NT;
    int kt   = (frag / NT) % KT;
    int r    = frag / (NT * KT);
    int k    = kt * 32 + (lane >> 4) * 16 + j;
    int n    = nt * 16 + (lane & 15);
    Wp[i] = (_Float16)W[((size_t)r * KDIM + k) * NOUT + n];
}

// ---------------- relation bucketing ----------------
__global__ void count_kernel(const int* __restrict__ edge_type, unsigned* __restrict__ cnt, int E) {
    int e = blockIdx.x * blockDim.x + threadIdx.x;
    if (e < E) atomicAdd(&cnt[edge_type[e] & (NREL - 1)], 1u);
}

__global__ void scan_kernel(const unsigned* __restrict__ cnt, unsigned* __restrict__ poff) {
    if (threadIdx.x == 0 && blockIdx.x == 0) {
        unsigned acc = 0;
        for (int r = 0; r < NREL; ++r) {
            poff[r] = acc;
            acc += (cnt[r] + 15u) & ~15u;   // pad each bucket to a 16-edge tile
        }
        poff[NREL] = acc;
    }
}

__global__ void scatter_kernel(const int* __restrict__ edge_type,
                               const unsigned* __restrict__ poff,
                               unsigned* __restrict__ cursor,
                               int* __restrict__ sorted, int E) {
    int e = blockIdx.x * blockDim.x + threadIdx.x;
    if (e >= E) return;
    int r = edge_type[e] & (NREL - 1);
    unsigned p = atomicAdd(&cursor[r], 1u);
    sorted[poff[r] + p] = e;
}

// ---------------- tiled relation GEMM + scatter-add ----------------
// One block = one 16-edge tile (single relation). 8 waves split the NOUT/16
// column tiles; K is swept 32 at a time with v_wmma_f32_16x16x32_f16.
template <int KDIM, int NOUT>
__global__ __launch_bounds__(256)
void rgcn_gemm_kernel(const _Float16* __restrict__ Xh,     // (N, KDIM) f16
                      const _Float16* __restrict__ Wp,     // packed weights
                      const int* __restrict__ sorted,
                      const int* __restrict__ edge_index,  // (2, E)
                      const int* __restrict__ edge_type,
                      const unsigned* __restrict__ poff,   // [17]
                      float* __restrict__ Out,             // (N, NOUT) f32, pre-zeroed
                      int E) {
    constexpr int KT = KDIM / 32;
    constexpr int NT = NOUT / 16;
    __shared__ _Float16 A[16 * KDIM];
    __shared__ int s_eid[16];
    __shared__ int s_dst[16];
    __shared__ int s_rel;

    const int tile = blockIdx.x;
    const unsigned total = poff[NREL];
    if ((unsigned)tile * 16u >= total) return;   // uniform exit

    const int tid = threadIdx.x;
    if (tid < 16) {
        int eid = sorted[tile * 16 + tid];
        s_eid[tid] = eid;
        s_dst[tid] = (eid >= 0) ? edge_index[E + eid] : -1;
        if (tid == 0) s_rel = edge_type[eid] & (NREL - 1); // row 0 always valid
    }
    __syncthreads();

    // gather 16 source rows (zeros for padding rows) into LDS, 16B chunks
    constexpr int CPR = KDIM / 8;
    for (int c = tid; c < 16 * CPR; c += 256) {
        int row = c / CPR, ch = c % CPR;
        int eid = s_eid[row];
        v8h val = {};
        if (eid >= 0) {
            int src = edge_index[eid];
            val = *(const v8h*)(Xh + (size_t)src * KDIM + ch * 8);
        }
        *(v8h*)(&A[row * KDIM + ch * 8]) = val;
    }
    __syncthreads();

    const int lane = tid & 31;
    const int wave = tid >> 5;
    const int hi   = lane >> 4;   // lane half selects K sub-range
    const int m    = lane & 15;
    const int r    = s_rel;

    for (int nt = wave; nt < NT; nt += 8) {
        v8f acc = {};
        for (int kt = 0; kt < KT; ++kt) {
            const int k0 = kt * 32;
            // A fragment: lanes 0-15 -> K k0+0..15 ; lanes 16-31 -> K k0+16..31
            v8h alo = *(const v8h*)(&A[m * KDIM + k0 + hi * 16]);
            v8h ahi = *(const v8h*)(&A[m * KDIM + k0 + hi * 16 + 8]);
            v16h a;
#pragma unroll
            for (int i = 0; i < 8; ++i) { a[i] = alo[i]; a[i + 8] = ahi[i]; }
            // B fragment: one coalesced 32B load per lane from packed weights
            const _Float16* bp = Wp + (((size_t)(r * KT + kt) * NT + nt) * 512 + lane * 16);
            v8h b0 = *(const v8h*)(bp);
            v8h b1 = *(const v8h*)(bp + 8);
            v16h b;
#pragma unroll
            for (int i = 0; i < 8; ++i) { b[i] = b0[i]; b[i + 8] = b1[i]; }
            acc = __builtin_amdgcn_wmma_f32_16x16x32_f16(
                /*neg_a=*/false, a, /*neg_b=*/false, b,
                /*c_mod=*/(short)0, acc, /*reuse_a=*/false, /*reuse_b=*/false);
        }
        // D layout: VGPR v -> row (hi*8+v), col = nt*16 + (lane&15)
        const int col = nt * 16 + m;
#pragma unroll
        for (int v = 0; v < 8; ++v) {
            int dst = s_dst[hi * 8 + v];
            if (dst >= 0) atomicAdd(&Out[(size_t)dst * NOUT + col], acc[v]);
        }
    }
}

// ---------------- LayerNorm + ReLU -> f16 (one wave32 per 256-wide row) ----
__global__ __launch_bounds__(256)
void ln_relu_kernel(const float* __restrict__ X1, const float* __restrict__ g,
                    const float* __restrict__ b, _Float16* __restrict__ X1h, int n) {
    int row = blockIdx.x * 8 + (threadIdx.x >> 5);
    if (row >= n) return;
    int lane = threadIdx.x & 31;
    const float* x = X1 + (size_t)row * HID_DIM;
    float v[8], s = 0.f;
#pragma unroll
    for (int i = 0; i < 8; ++i) { v[i] = x[lane + i * 32]; s += v[i]; }
#pragma unroll
    for (int off = 16; off >= 1; off >>= 1) s += __shfl_xor(s, off, 32);
    float mu = s * (1.f / HID_DIM), q = 0.f;
#pragma unroll
    for (int i = 0; i < 8; ++i) { float d = v[i] - mu; q += d * d; }
#pragma unroll
    for (int off = 16; off >= 1; off >>= 1) q += __shfl_xor(q, off, 32);
    float inv = rsqrtf(q * (1.f / HID_DIM) + 1e-5f);
    _Float16* y = X1h + (size_t)row * HID_DIM;
#pragma unroll
    for (int i = 0; i < 8; ++i) {
        int c = lane + i * 32;
        float t = (v[i] - mu) * inv * g[c] + b[c];
        y[c] = (_Float16)fmaxf(t, 0.f);
    }
}

// ---------------- LayerNorm + residual -> out (128-wide rows) --------------
__global__ __launch_bounds__(256)
void ln_add_kernel(const float* __restrict__ X2, const float* __restrict__ g,
                   const float* __restrict__ b, const float* __restrict__ emb,
                   const int* __restrict__ node_ids, float* __restrict__ out, int n) {
    int row = blockIdx.x * 8 + (threadIdx.x >> 5);
    if (row >= n) return;
    int lane = threadIdx.x & 31;
    const float* x = X2 + (size_t)row * EMB_DIM;
    float v[4], s = 0.f;
#pragma unroll
    for (int i = 0; i < 4; ++i) { v[i] = x[lane + i * 32]; s += v[i]; }
#pragma unroll
    for (int off = 16; off >= 1; off >>= 1) s += __shfl_xor(s, off, 32);
    float mu = s * (1.f / EMB_DIM), q = 0.f;
#pragma unroll
    for (int i = 0; i < 4; ++i) { float d = v[i] - mu; q += d * d; }
#pragma unroll
    for (int off = 16; off >= 1; off >>= 1) q += __shfl_xor(q, off, 32);
    float inv = rsqrtf(q * (1.f / EMB_DIM) + 1e-5f);
    const float* res = emb + (size_t)node_ids[row] * EMB_DIM;
    float* y = out + (size_t)row * EMB_DIM;
#pragma unroll
    for (int i = 0; i < 4; ++i) {
        int c = lane + i * 32;
        y[c] = (v[i] - mu) * inv * g[c] + b[c] + res[c];
    }
}

// ---------------------------------------------------------------------------
static inline size_t align256(size_t x) { return (x + 255) & ~(size_t)255; }

extern "C" void kernel_launch(void* const* d_in, const int* in_sizes, int n_in,
                              void* d_out, int out_size, void* d_ws, size_t ws_size,
                              hipStream_t stream) {
    const int*   node_ids   = (const int*)d_in[0];
    const int*   edge_index = (const int*)d_in[1];
    const int*   edge_type  = (const int*)d_in[2];
    const float* emb        = (const float*)d_in[3];
    const float* W1         = (const float*)d_in[4];
    const float* W2         = (const float*)d_in[5];
    const float* g1         = (const float*)d_in[6];
    const float* b1         = (const float*)d_in[7];
    const float* g2         = (const float*)d_in[8];
    const float* b2         = (const float*)d_in[9];
    float* out = (float*)d_out;

    const int N = in_sizes[0];
    const int E = in_sizes[2];

    // ---- workspace layout ----
    char* ws = (char*)d_ws;
    size_t off = 0;
    unsigned* cnt    = (unsigned*)(ws + off); off += 64;            // 16 u32
    unsigned* cursor = (unsigned*)(ws + off); off += 64;            // 16 u32
    unsigned* poff   = (unsigned*)(ws + off); off = align256(off + 17 * 4);
    int*      sorted = (int*)(ws + off);      off = align256(off + (size_t)(E + 256) * 4);
    _Float16* Xh     = (_Float16*)(ws + off); off = align256(off + (size_t)N * EMB_DIM * 2);
    _Float16* X1h    = (_Float16*)(ws + off); off = align256(off + (size_t)N * HID_DIM * 2);
    _Float16* Wp1    = (_Float16*)(ws + off); off = align256(off + (size_t)NREL * EMB_DIM * HID_DIM * 2);
    _Float16* Wp2    = (_Float16*)(ws + off); off = align256(off + (size_t)NREL * HID_DIM * EMB_DIM * 2);
    float*    X1     = (float*)(ws + off);    off = align256(off + (size_t)N * HID_DIM * 4);
    float*    X2     = (float*)(ws + off);    off = align256(off + (size_t)N * EMB_DIM * 4);

    // ---- init (deterministic every call) ----
    fill_u32_kernel<<<1, 64, 0, stream>>>((unsigned*)cnt, 0u, 64 + 17); // cnt+cursor+poff
    fill_u32_kernel<<<1024, 256, 0, stream>>>((unsigned*)sorted, 0xFFFFFFFFu, (size_t)(E + 256));
    fill_u32_kernel<<<4096, 256, 0, stream>>>((unsigned*)X1, 0u, (size_t)N * (HID_DIM + EMB_DIM)); // X1+X2 contiguous

    // ---- f16 conversions / weight packing ----
    {
        int total = N * EMB_DIM;
        cvt_x_kernel<<<(total + 255) / 256, 256, 0, stream>>>(emb, node_ids, Xh, total);
    }
    {
        constexpr int T1 = NREL * (EMB_DIM / 32) * (HID_DIM / 16) * 512;
        pack_w_kernel<EMB_DIM, HID_DIM><<<(T1 + 255) / 256, 256, 0, stream>>>(W1, Wp1);
        constexpr int T2 = NREL * (HID_DIM / 32) * (EMB_DIM / 16) * 512;
        pack_w_kernel<HID_DIM, EMB_DIM><<<(T2 + 255) / 256, 256, 0, stream>>>(W2, Wp2);
    }

    // ---- bucket edges by relation (padded to 16-edge tiles) ----
    count_kernel<<<(E + 255) / 256, 256, 0, stream>>>(edge_type, cnt, E);
    scan_kernel<<<1, 32, 0, stream>>>(cnt, poff);
    scatter_kernel<<<(E + 255) / 256, 256, 0, stream>>>(edge_type, poff, cursor, sorted, E);

    const int maxTiles = E / 16 + NREL;   // upper bound on padded tile count

    // ---- layer 1: propagate (WMMA) + LN + ReLU ----
    rgcn_gemm_kernel<EMB_DIM, HID_DIM><<<maxTiles, 256, 0, stream>>>(
        Xh, Wp1, sorted, edge_index, edge_type, poff, X1, E);
    ln_relu_kernel<<<(N + 7) / 8, 256, 0, stream>>>(X1, g1, b1, X1h, N);

    // ---- layer 2: propagate (WMMA) + LN + residual ----
    rgcn_gemm_kernel<HID_DIM, EMB_DIM><<<maxTiles, 256, 0, stream>>>(
        X1h, Wp2, sorted, edge_index, edge_type, poff, X2, E);
    ln_add_kernel<<<(N + 7) / 8, 256, 0, stream>>>(X2, g2, b2, emb, node_ids, out, N);
}
